// ReLlamaAttention_77781857730645
// MI455X (gfx1250) — matrix operand order repaired and checked by reading, not compiled, not run on hardware
//
#include <hip/hip_runtime.h>
#include <hip/hip_bf16.h>

// ---------------------------------------------------------------------------
// ReLlama retrieval attention for MI455X (gfx1250, wave32, WMMA + TDM).
// bf16 in / f32 accumulate via v_wmma_f32_16x16x32_bf16; GEMM tiles staged
// into LDS with tensor_load_to_lds (Tensor Data Mover), double-buffered.
// ---------------------------------------------------------------------------

typedef __attribute__((ext_vector_type(16))) __bf16 v16bf;
typedef __attribute__((ext_vector_type(8)))  __bf16 v8bf;
typedef __attribute__((ext_vector_type(8)))  float  v8f;
typedef __attribute__((ext_vector_type(4)))  unsigned int u32x4;
typedef __attribute__((ext_vector_type(8)))  int i32x8;
typedef __attribute__((ext_vector_type(4)))  int i32x4;

#define Hq    16
#define HKVh  4
#define Dh    128
#define Sq    2048
#define HIDs  2048
#define RLENs 2048

static __device__ __forceinline__ v8f wmma_bf16(v16bf a, v16bf b, v8f c) {
  return __builtin_amdgcn_wmma_f32_16x16x32_bf16(false, a, false, b, (short)0, c,
                                                 false, false);
}

// ------------------------- Tensor Data Mover helper ------------------------
// 2D tile load (bf16 elements): tile_w x tile_h from a row-major tensor with
// row stride `row_stride` (elements). D# packing per CDNA5 ISA ch.8.

static __device__ __forceinline__ unsigned lds_off(const void* p) {
  // generic->LDS aperture: low 32 bits are the LDS byte offset
  return (unsigned)(unsigned long long)(uintptr_t)p;
}

static __device__ __forceinline__ void tdm_load_2d(
    unsigned ldsAddr, const void* gptr, unsigned tile_w, unsigned tile_h,
    unsigned tensor_w, unsigned tensor_h, unsigned row_stride) {
  unsigned long long ga = (unsigned long long)(uintptr_t)gptr;
  u32x4 g0;
  g0[0] = 1u;                                     // count=1, user descriptor
  g0[1] = ldsAddr;                                // lds_addr (bytes)
  g0[2] = (unsigned)(ga & 0xFFFFFFFFu);           // global_addr[31:0]
  g0[3] = (unsigned)((ga >> 32) & 0x01FFFFFFu)    // global_addr[56:32]
          | 0x80000000u;                          // type=2 ("image")
  i32x8 g1;
  g1[0] = (int)(1u << 16);                        // data_size=1 -> 2 bytes
  g1[1] = (int)((tensor_w & 0xFFFFu) << 16);      // tensor_dim0[15:0]
  g1[2] = (int)(((tensor_w >> 16) & 0xFFFFu) |    // tensor_dim0[31:16]
                ((tensor_h & 0xFFFFu) << 16));    // tensor_dim1[15:0]
  g1[3] = (int)(((tensor_h >> 16) & 0xFFFFu) |    // tensor_dim1[31:16]
                ((tile_w & 0xFFFFu) << 16));      // tile_dim0
  g1[4] = (int)(tile_h & 0xFFFFu);                // tile_dim1 (tile_dim2=0)
  g1[5] = (int)row_stride;                        // tensor_dim0_stride[31:0]
  g1[6] = 0;
  g1[7] = 0;
  i32x4 z4 = {0, 0, 0, 0};
#if defined(__clang_major__) && (__clang_major__ >= 23)
  i32x8 z8 = {0, 0, 0, 0, 0, 0, 0, 0};
  __builtin_amdgcn_tensor_load_to_lds(g0, g1, z4, z4, z8, 0);
#else
  __builtin_amdgcn_tensor_load_to_lds(g0, g1, z4, z4, 0);
#endif
}

// ------------------------------ layout kernels -----------------------------

__global__ void cvt_bf16_kernel(const float* __restrict__ in,
                                __bf16* __restrict__ out, int n) {
  int i = blockIdx.x * blockDim.x + threadIdx.x;
  if (i < n) out[i] = (__bf16)in[i];
}

// Wt[n*K + k] = bf16(W[k*N + n])   (W is KxN row-major)
__global__ void transpose_w_kernel(const float* __restrict__ W,
                                   __bf16* __restrict__ Wt, int K, int N) {
  int i = blockIdx.x * blockDim.x + threadIdx.x;
  if (i >= K * N) return;
  int k = i / N, n = i - k * N;
  Wt[(size_t)n * K + k] = (__bf16)W[i];
}

// Qf[s][h*D+d] -> rope -> *scale -> Qb[h][s][d]
__global__ void rope_q_kernel(const float* __restrict__ Qf,
                              const float* __restrict__ cosb,
                              const float* __restrict__ sinb,
                              __bf16* __restrict__ Qb) {
  int idx = blockIdx.x * blockDim.x + threadIdx.x;  // < S*H*D
  int d = idx & (Dh - 1);
  int rem = idx >> 7;          // s*H + h
  int hh = rem & (Hq - 1);
  int s = rem >> 4;
  float x = Qf[idx];
  float c = cosb[s * Dh + d], sn = sinb[s * Dh + d];
  size_t rowbase = (size_t)s * HIDs + hh * Dh;
  float rot = (d < 64) ? -Qf[rowbase + d + 64] : Qf[rowbase + d - 64];
  float val = (x * c + rot * sn) * 0.08838834764831845f;  // 1/sqrt(128)
  Qb[((size_t)hh * Sq + s) * Dh + d] = (__bf16)val;
}

// Kf[s][kv*D+d] -> rope -> Kb[kv][s][d]
__global__ void rope_k_kernel(const float* __restrict__ Kf,
                              const float* __restrict__ cosb,
                              const float* __restrict__ sinb,
                              __bf16* __restrict__ Kb) {
  int idx = blockIdx.x * blockDim.x + threadIdx.x;  // < S*HKV*D
  int d = idx & (Dh - 1);
  int rem = idx >> 7;          // s*HKV + kv
  int kvh = rem & (HKVh - 1);
  int s = rem >> 2;
  float x = Kf[idx];
  float c = cosb[s * Dh + d], sn = sinb[s * Dh + d];
  size_t rowbase = (size_t)s * (HKVh * Dh) + kvh * Dh;
  float rot = (d < 64) ? -Kf[rowbase + d + 64] : Kf[rowbase + d - 64];
  Kb[((size_t)kvh * Sq + s) * Dh + d] = (__bf16)(x * c + rot * sn);
}

// Vf[s][kv*D+d] -> Vt[kv][d][s]
__global__ void vtrans_kernel(const float* __restrict__ Vf,
                              __bf16* __restrict__ Vt) {
  int idx = blockIdx.x * blockDim.x + threadIdx.x;  // < S*HKV*D
  int d = idx & (Dh - 1);
  int rem = idx >> 7;
  int kvh = rem & (HKVh - 1);
  int s = rem >> 2;
  Vt[((size_t)kvh * Dh + d) * Sq + s] = (__bf16)Vf[idx];
}

// rv[kv][r][d] -> rVt[kv][d][r]
__global__ void rv_trans_kernel(const float* __restrict__ rv,
                                __bf16* __restrict__ rVt) {
  int idx = blockIdx.x * blockDim.x + threadIdx.x;  // < HKV*RLEN*D
  int d = idx & (Dh - 1);
  int rem = idx >> 7;
  int r = rem & (RLENs - 1);
  int kvh = rem >> 11;
  rVt[((size_t)kvh * Dh + d) * RLENs + r] = (__bf16)rv[idx];
}

// ------------------------------- bf16 GEMM ---------------------------------
// C[M,N](f32) = A[M,K](bf16 row-major) x B given as Bt[N,K](bf16 row-major).
// 256 threads (8 waves) per block; block computes a 128x128 C tile.
// A/B 128x64 tiles staged into LDS by the Tensor Data Mover, double-buffered:
// DMA for step k+1 overlaps WMMA compute on step k. Each wave owns a 32x64
// strip = 2x4 accumulator tiles, 16 WMMAs per 64-deep K step fed by ds_load.

#define BM 128
#define BN 128
#define BK 64

__global__ void __launch_bounds__(256)
gemm_bf16_kernel(const __bf16* __restrict__ A, const __bf16* __restrict__ Bt,
                 float* __restrict__ C, int M, int N, int K) {
  __shared__ __align__(32) __bf16 Atile[2][BM][BK];
  __shared__ __align__(32) __bf16 Btile[2][BN][BK];

  int wave = threadIdx.x >> 5;
  int lane = threadIdx.x & 31;
  int col = lane & 15, half = lane >> 4;
  int nBlkN = N >> 7;
  int bm = blockIdx.x / nBlkN;
  int bn = blockIdx.x - bm * nBlkN;
  int waveM = wave >> 1;   // 0..3 -> 32 rows each
  int waveN = wave & 1;    // 0..1 -> 64 cols each

  v8f acc[2][4];
#pragma unroll
  for (int mt = 0; mt < 2; ++mt)
#pragma unroll
    for (int j = 0; j < 4; ++j)
#pragma unroll
      for (int r = 0; r < 8; ++r) acc[mt][j][r] = 0.f;

  const __bf16* Ablk = A + (size_t)bm * BM * K;
  const __bf16* Bblk = Bt + (size_t)bn * BN * K;
  int nK = K / BK;

  if (wave == 0) {  // wave-uniform: prologue DMA of tile pair 0
    tdm_load_2d(lds_off(&Atile[0][0][0]), Ablk, BK, BM, (unsigned)K,
                (unsigned)M, (unsigned)K);
    tdm_load_2d(lds_off(&Btile[0][0][0]), Bblk, BK, BN, (unsigned)K,
                (unsigned)N, (unsigned)K);
  }

  for (int kb = 0; kb < nK; ++kb) {
    int cur = kb & 1;
    if (wave == 0) __builtin_amdgcn_s_wait_tensorcnt(0);
    __syncthreads();  // tile[cur] ready; everyone done reading tile[cur^1]
    if (wave == 0 && (kb + 1) < nK) {  // prefetch next tile pair via TDM
      tdm_load_2d(lds_off(&Atile[cur ^ 1][0][0]), Ablk + (size_t)(kb + 1) * BK,
                  BK, BM, (unsigned)K, (unsigned)M, (unsigned)K);
      tdm_load_2d(lds_off(&Btile[cur ^ 1][0][0]), Bblk + (size_t)(kb + 1) * BK,
                  BK, BN, (unsigned)K, (unsigned)N, (unsigned)K);
    }
#pragma unroll
    for (int ks = 0; ks < BK; ks += 32) {
      union { v16bf v; v8bf h[2]; } a[2];
#pragma unroll
      for (int mt = 0; mt < 2; ++mt) {
        const __bf16* ap = &Atile[cur][waveM * 32 + mt * 16 + col][ks + half * 16];
        a[mt].h[0] = *(const v8bf*)ap;
        a[mt].h[1] = *(const v8bf*)(ap + 8);
      }
#pragma unroll
      for (int j = 0; j < 4; ++j) {
        const __bf16* bp =
            &Btile[cur][waveN * 64 + j * 16 + col][ks + half * 16];
        v16bf b = *(const v16bf*)bp;
#pragma unroll
        for (int mt = 0; mt < 2; ++mt)
          acc[mt][j] = wmma_bf16(a[mt].v, b, acc[mt][j]);
      }
    }
  }

#pragma unroll
  for (int mt = 0; mt < 2; ++mt)
#pragma unroll
    for (int j = 0; j < 4; ++j)
#pragma unroll
      for (int r = 0; r < 8; ++r)
        C[(size_t)(bm * BM + waveM * 32 + mt * 16 + r + half * 8) * N +
          bn * BN + waveN * 64 + j * 16 + col] = acc[mt][j][r];
}

// ------------------------------ attention ----------------------------------
// Per wave: one (head, 16-query tile). Scores computed transposed
// (S^T = K tile (A) x Q^T (B)) so each lane owns one query column.
// O^T accumulated as 8 d-chunk tiles of 16x16.

static __device__ __forceinline__ void attn_block(
    const __bf16* __restrict__ Krow,   // K rows for this kv head ([key][d])
    const __bf16* __restrict__ Vrow,   // V^T rows for this kv head ([d][key])
    int keyBase, int q, int col, int half, bool domask, const v16bf* qf,
    __bf16* Pw, float& m, float& l, v8f* acc, size_t vld) {
  v8f st[2];
#pragma unroll
  for (int sub = 0; sub < 2; ++sub) {
    v8f s;
#pragma unroll
    for (int r = 0; r < 8; ++r) s[r] = 0.f;
    const __bf16* kr =
        Krow + (size_t)(keyBase + sub * 16 + col) * Dh + half * 16;
#pragma unroll
    for (int dc = 0; dc < 4; ++dc) {
      union { v16bf v; v8bf h[2]; } A;
      A.h[0] = *(const v8bf*)(kr + dc * 32);
      A.h[1] = *(const v8bf*)(kr + dc * 32 + 8);
      s = wmma_bf16(A.v, qf[dc], s);
    }
    st[sub] = s;
  }
  float sv[16];
  float tmax = -3.0e38f;
#pragma unroll
  for (int sub = 0; sub < 2; ++sub)
#pragma unroll
    for (int r = 0; r < 8; ++r) {
      int kk = sub * 16 + r + half * 8;       // key within 32-block
      float sval = st[sub][r];
      if (domask && (keyBase + kk) > q) sval = -3.0e38f;
      sv[sub * 8 + r] = sval;
      tmax = fmaxf(tmax, sval);
    }
  tmax = fmaxf(tmax, __shfl_xor(tmax, 16, 32));
  float mnew = fmaxf(m, tmax);
  float alpha = __expf(m - mnew);
  float rsum = 0.f;
#pragma unroll
  for (int sub = 0; sub < 2; ++sub)
#pragma unroll
    for (int r = 0; r < 8; ++r) {
      float p = __expf(sv[sub * 8 + r] - mnew);
      rsum += p;
      Pw[col * 32 + sub * 16 + r + half * 8] = (__bf16)p;
    }
  rsum += __shfl_xor(rsum, 16, 32);
  l = l * alpha + rsum;
  m = mnew;
#pragma unroll
  for (int t = 0; t < 8; ++t)
#pragma unroll
    for (int r = 0; r < 8; ++r) acc[t][r] *= alpha;

  // LDS stores (ds_store_b16) must land before the fragment read-back.
  asm volatile("s_wait_dscnt 0" ::: "memory");
  v16bf pb = *(const v16bf*)(Pw + col * 32 + half * 16);
#pragma unroll
  for (int t = 0; t < 8; ++t) {
    const __bf16* vr =
        Vrow + (size_t)(t * 16 + col) * vld + keyBase + half * 16;
    union { v16bf v; v8bf h[2]; } A;
    A.h[0] = *(const v8bf*)(vr);
    A.h[1] = *(const v8bf*)(vr + 8);
    acc[t] = wmma_bf16(A.v, pb, acc[t]);
  }
}

__global__ void __launch_bounds__(128)
attn_kernel(const __bf16* __restrict__ Qb, const __bf16* __restrict__ Kb,
            const __bf16* __restrict__ Vt, const __bf16* __restrict__ rKb,
            const __bf16* __restrict__ rVt, __bf16* __restrict__ attnB) {
  __shared__ __align__(32) __bf16 Pbuf[4][16 * 32];
  int wave = threadIdx.x >> 5;
  int lane = threadIdx.x & 31;
  int gw = blockIdx.x * 4 + wave;
  int h = gw >> 7;               // 128 query tiles per head
  int qTile = gw & 127;
  int kvh = h >> 2;              // G = 4
  int col = lane & 15, half = lane >> 4;
  int qStart = qTile * 16;
  int q = qStart + col;
  __bf16* Pw = &Pbuf[wave][0];

  v16bf qf[4];
  const __bf16* qrow = Qb + ((size_t)h * Sq + q) * Dh + half * 16;
#pragma unroll
  for (int dc = 0; dc < 4; ++dc) qf[dc] = *(const v16bf*)(qrow + dc * 32);

  float m = -3.0e38f, l = 0.f;
  v8f acc[8];
#pragma unroll
  for (int t = 0; t < 8; ++t)
#pragma unroll
    for (int r = 0; r < 8; ++r) acc[t][r] = 0.f;

  const __bf16* Kh = Kb + (size_t)kvh * Sq * Dh;
  const __bf16* Vh = Vt + (size_t)kvh * Dh * Sq;
  int nb = (qStart + 16 + 31) >> 5;
  for (int b = 0; b < nb; ++b) {
    int keyBase = b * 32;
    bool mask = (keyBase + 31) > qStart;
    attn_block(Kh, Vh, keyBase, q, col, half, mask, qf, Pw, m, l, acc,
               (size_t)Sq);
  }
  int chunk = qStart >> 6;       // CHUNK = 64
  if (chunk >= 1) {              // wave-uniform
    int blk = chunk - 1;         // == clip(chunk-1, 0, R-1)
    const __bf16* Krh = rKb + (size_t)kvh * RLENs * Dh;
    const __bf16* Vrh = rVt + (size_t)kvh * Dh * RLENs;
    attn_block(Krh, Vrh, blk * 64, q, col, half, false, qf, Pw, m, l, acc,
               (size_t)RLENs);
    attn_block(Krh, Vrh, blk * 64 + 32, q, col, half, false, qf, Pw, m, l,
               acc, (size_t)RLENs);
  }
  float inv = 1.0f / l;
#pragma unroll
  for (int t = 0; t < 8; ++t)
#pragma unroll
    for (int r = 0; r < 8; ++r) {
      int d = t * 16 + r + half * 8;
      attnB[(size_t)q * (Hq * Dh) + h * Dh + d] = (__bf16)(acc[t][r] * inv);
    }
}

// ------------------------------- launcher ----------------------------------

extern "C" void kernel_launch(void* const* d_in, const int* in_sizes, int n_in,
                              void* d_out, int out_size, void* d_ws,
                              size_t ws_size, hipStream_t stream) {
  const float* hidden = (const float*)d_in[0];
  const float* cosb   = (const float*)d_in[1];
  const float* sinb   = (const float*)d_in[2];
  const float* rk     = (const float*)d_in[3];
  const float* rv     = (const float*)d_in[4];
  const float* Wq     = (const float*)d_in[5];
  const float* Wk     = (const float*)d_in[6];
  const float* Wv     = (const float*)d_in[7];
  const float* Wo     = (const float*)d_in[8];
  float* out = (float*)d_out;

  char* ws = (char*)d_ws;
  size_t off = 0;
  auto alloc = [&](size_t bytes) -> void* {
    void* p = ws + off;
    off += (bytes + 255) & ~(size_t)255;
    return p;
  };
  const size_t SZ = 2048;
  __bf16* Xb   = (__bf16*)alloc(SZ * 2048 * 2);          // hidden bf16
  __bf16* WqT  = (__bf16*)alloc(2048 * 2048 * 2);
  __bf16* WkT  = (__bf16*)alloc(512 * 2048 * 2);
  __bf16* WvT  = (__bf16*)alloc(512 * 2048 * 2);
  __bf16* WoT  = (__bf16*)alloc(2048 * 2048 * 2);
  float*  Qf   = (float*)alloc(SZ * 2048 * 4);
  float*  Kf   = (float*)alloc(SZ * 512 * 4);
  float*  Vf   = (float*)alloc(SZ * 512 * 4);
  __bf16* Qb   = (__bf16*)alloc((size_t)Hq * Sq * Dh * 2);
  __bf16* KbB  = (__bf16*)alloc((size_t)HKVh * Sq * Dh * 2);
  __bf16* VtB  = (__bf16*)alloc((size_t)HKVh * Dh * Sq * 2);
  __bf16* rKb  = (__bf16*)alloc((size_t)HKVh * RLENs * Dh * 2);
  __bf16* rVt  = (__bf16*)alloc((size_t)HKVh * Dh * RLENs * 2);
  __bf16* attnB = (__bf16*)alloc(SZ * 2048 * 2);

  const int T = 256;
  // 1) precision / layout conversions
  cvt_bf16_kernel<<<(2048 * 2048) / T, T, 0, stream>>>(hidden, Xb,
                                                       2048 * 2048);
  transpose_w_kernel<<<(2048 * 2048) / T, T, 0, stream>>>(Wq, WqT, 2048, 2048);
  transpose_w_kernel<<<(2048 * 512) / T, T, 0, stream>>>(Wk, WkT, 2048, 512);
  transpose_w_kernel<<<(2048 * 512) / T, T, 0, stream>>>(Wv, WvT, 2048, 512);
  transpose_w_kernel<<<(2048 * 2048) / T, T, 0, stream>>>(Wo, WoT, 2048, 2048);
  cvt_bf16_kernel<<<(HKVh * RLENs * Dh) / T, T, 0, stream>>>(
      rk, rKb, HKVh * RLENs * Dh);
  rv_trans_kernel<<<(HKVh * RLENs * Dh) / T, T, 0, stream>>>(rv, rVt);

  // 2) QKV projections (TDM-staged bf16 WMMA GEMM, f32 out)
  gemm_bf16_kernel<<<16 * 16, 256, 0, stream>>>(Xb, WqT, Qf, 2048, 2048, 2048);
  gemm_bf16_kernel<<<16 * 4, 256, 0, stream>>>(Xb, WkT, Kf, 2048, 512, 2048);
  gemm_bf16_kernel<<<16 * 4, 256, 0, stream>>>(Xb, WvT, Vf, 2048, 512, 2048);

  // 3) RoPE + attention-friendly layouts
  rope_q_kernel<<<(2048 * 2048) / T, T, 0, stream>>>(Qf, cosb, sinb, Qb);
  rope_k_kernel<<<(2048 * 512) / T, T, 0, stream>>>(Kf, cosb, sinb, KbB);
  vtrans_kernel<<<(2048 * 512) / T, T, 0, stream>>>(Vf, VtB);

  // 4) fused causal + retrieval flash attention
  attn_kernel<<<(Hq * 128) / 4, 128, 0, stream>>>(Qb, KbB, VtB, rKb, rVt,
                                                  attnB);

  // 5) output projection -> fp32 result
  gemm_bf16_kernel<<<16 * 16, 256, 0, stream>>>(attnB, WoT, out, 2048, 2048,
                                                2048);
}
